// RticCompositionModule_53171695124669
// MI455X (gfx1250) — matrix operand order repaired and checked by reading, not compiled, not run on hardware
//
#include <hip/hip_runtime.h>
#include <hip/hip_bf16.h>
#include <stdint.h>

typedef __bf16 bf16_t;
typedef __attribute__((ext_vector_type(16))) __bf16 v16bf;
typedef __attribute__((ext_vector_type(8)))  __bf16 v8bf;
typedef __attribute__((ext_vector_type(4)))  __bf16 v4bf;
typedef __attribute__((ext_vector_type(2)))  __bf16 v2bf;
typedef __attribute__((ext_vector_type(8)))  float  v8f;
typedef __attribute__((ext_vector_type(4)))  float  v4f;

#define E_DIM 4096
#define H_DIM 2048
#define BSZ   32
#define KSPL  8      // K-split factor (deterministic partials, no atomics)
#define NTB   128    // N columns per block (each of 8 waves: 2 x 16-col WMMA tiles)
#define LDK   40     // LDS row stride in bf16 (80 B, keeps 16B-aligned b128 chunks)
#define BN_EPS 1e-5f

// native f32 -> bf16 (RNE); lowers to the hardware convert (v_cvt_pk_bf16_f32
// when pair-packed) instead of a ~10-VALU bit-twiddle sequence
__device__ __forceinline__ bf16_t f2bf(float f) { return (bf16_t)f; }

// ---------------------------------------------------------------------------
// Prep: xn = relu(BN(concat(image, broadcast(text)))) as bf16 [32, 8192].
// Text half of BN collapses exactly to beta (variance of identical rows == 0).
// ---------------------------------------------------------------------------
__global__ __launch_bounds__(256) void prep_xn_kernel(
    const float* __restrict__ img, const float* __restrict__ bn_g,
    const float* __restrict__ bn_b, bf16_t* __restrict__ xn)
{
    const int c = blockIdx.x * blockDim.x + threadIdx.x;   // 0..8191
    if (c < E_DIM) {
        float v[BSZ]; float mu = 0.f;
        #pragma unroll
        for (int r = 0; r < BSZ; ++r) { v[r] = img[r * E_DIM + c]; mu += v[r]; }
        mu *= (1.f / BSZ);
        float var = 0.f;
        #pragma unroll
        for (int r = 0; r < BSZ; ++r) { float d = v[r] - mu; var += d * d; }
        var *= (1.f / BSZ);
        const float rs = rsqrtf(var + BN_EPS);
        const float g = bn_g[c], b = bn_b[c];
        #pragma unroll
        for (int r = 0; r < BSZ; ++r)
            xn[r * (2 * E_DIM) + c] = f2bf(fmaxf((v[r] - mu) * rs * g + b, 0.f));
    } else {
        const bf16_t t = f2bf(fmaxf(bn_b[c], 0.f));
        #pragma unroll
        for (int r = 0; r < BSZ; ++r) xn[r * (2 * E_DIM) + c] = t;
    }
}

// ---------------------------------------------------------------------------
// GEMM partial: part[ks][32][N] = A[32, Kslice] @ W[Kslice, N]
//   A bf16 row-major; W fp32 row-major, converted to bf16 in-register.
//   256 thr = 8 waves; block tile [32 x 128]; K-step 32; double-buffered LDS;
//   one barrier per K-step; 2 x v_wmma_f32_16x16x32_bf16 per wave per step.
//   N is a template constant so partial-store strides become immediates.
// ---------------------------------------------------------------------------
template <int N>
__global__ __launch_bounds__(256) void gemm_bf16_partial_kernel(
    const bf16_t* __restrict__ A, const float* __restrict__ W,
    float* __restrict__ part, int Ktot)
{
    __shared__ __align__(16) bf16_t sA[2][32 * LDK];    // [buf][m][k]
    __shared__ __align__(16) bf16_t sB[2][NTB * LDK];   // [buf][n][k] (transposed)

    const int tid   = threadIdx.x;
    const int wave  = tid >> 5;
    const int lane  = tid & 31;
    const int mtile = wave >> 2;        // 0..1 -> M rows 0-15 / 16-31
    const int npair = wave & 3;         // 0..3 -> 32-col slice (two WMMA tiles)
    const int mlow  = lane & 15;
    const int khalf = lane >> 4;

    const int c0     = blockIdx.x * NTB;
    const int ks     = blockIdx.y;
    const int Kblk   = Ktot / KSPL;
    const int k0base = ks * Kblk;
    const int nsteps = Kblk / 32;

    // cooperative load mappings
    const int a_m = tid >> 3;            // 0..31
    const int a_k = (tid & 7) * 4;       // 0..28
    const int b_k = (tid >> 4) * 2;      // 0..30  (pair of K rows)
    const int b_c = (tid & 15) * 8;      // 0..120 (8 columns)

    const bf16_t* aSrc = &A[(size_t)a_m * Ktot + k0base + a_k];
    const float*  wSrc = &W[(size_t)(k0base + b_k) * N + (c0 + b_c)];

    v8f acc0 = {}, acc1 = {};

    // staged tile registers
    v4bf aReg;
    v4f  b00, b01, b10, b11;

    auto loadTile = [&](int s) {
        const size_t wo = (size_t)s * 32 * N;
        aReg = *(const v4bf*)(aSrc + s * 32);
        b00  = __builtin_nontemporal_load((const v4f*)(wSrc + wo));
        b01  = __builtin_nontemporal_load((const v4f*)(wSrc + wo + 4));
        b10  = __builtin_nontemporal_load((const v4f*)(wSrc + wo + N));
        b11  = __builtin_nontemporal_load((const v4f*)(wSrc + wo + N + 4));
    };
    auto storeTile = [&](int buf) {
        *(v4bf*)&sA[buf][a_m * LDK + a_k] = aReg;
        #pragma unroll
        for (int i = 0; i < 4; ++i) {
            v2bf p; p.x = f2bf(b00[i]); p.y = f2bf(b10[i]);
            *(v2bf*)&sB[buf][(b_c + i) * LDK + b_k] = p;
            v2bf q; q.x = f2bf(b01[i]); q.y = f2bf(b11[i]);
            *(v2bf*)&sB[buf][(b_c + 4 + i) * LDK + b_k] = q;
        }
    };

    loadTile(0);
    storeTile(0);
    __syncthreads();

    const int aRow = (mtile * 16 + mlow) * LDK;
    const int bRow0 = (npair * 32 + mlow) * LDK;
    const int bRow1 = (npair * 32 + 16 + mlow) * LDK;

    for (int s = 0; s < nsteps; ++s) {
        const int buf = s & 1;
        if (s + 1 < nsteps) {
            loadTile(s + 1);                                  // overlap with WMMA
            __builtin_prefetch(wSrc + (size_t)(s + 2) * 32 * N, 0, 1);
        }
        // WMMA operands in contiguous k order (k = 16*khalf + s), same for A and B (ds_load_b128 x2 each)
        v8bf a0 = *(const v8bf*)&sA[buf][aRow + khalf * 16];
        v8bf a1 = *(const v8bf*)&sA[buf][aRow + khalf * 16 + 8];
        v8bf p0 = *(const v8bf*)&sB[buf][bRow0 + khalf * 16];
        v8bf p1 = *(const v8bf*)&sB[buf][bRow0 + khalf * 16 + 8];
        v8bf q0 = *(const v8bf*)&sB[buf][bRow1 + khalf * 16];
        v8bf q1 = *(const v8bf*)&sB[buf][bRow1 + khalf * 16 + 8];
        v16bf av = __builtin_shufflevector(a0, a1, 0,1,2,3,4,5,6,7,8,9,10,11,12,13,14,15);
        v16bf bv = __builtin_shufflevector(p0, p1, 0,1,2,3,4,5,6,7,8,9,10,11,12,13,14,15);
        v16bf cv = __builtin_shufflevector(q0, q1, 0,1,2,3,4,5,6,7,8,9,10,11,12,13,14,15);
        acc0 = __builtin_amdgcn_wmma_f32_16x16x32_bf16(
                   false, av, false, bv, (short)0, acc0, false, false);
        acc1 = __builtin_amdgcn_wmma_f32_16x16x32_bf16(
                   false, av, false, cv, (short)0, acc1, false, false);
        if (s + 1 < nsteps) storeTile(buf ^ 1);
        __syncthreads();
    }

    // C layout: lanes 0-15 -> M=r, lanes 16-31 -> M=8+r; N = lane%16
    const int m0  = mtile * 16 + khalf * 8;
    const int col = c0 + npair * 32 + mlow;
    float* p = &part[((size_t)ks * BSZ + m0) * N + col];
    #pragma unroll
    for (int r = 0; r < 8; ++r) {
        p[(size_t)r * N]      = acc0[r];   // compile-time stride -> imm offsets
        p[(size_t)r * N + 16] = acc1[r];
    }
}

// ---------------------------------------------------------------------------
// Epilogues (sum K-split partials deterministically, then fuse pointwise ops)
// ---------------------------------------------------------------------------
__global__ __launch_bounds__(256) void epi_bias_kernel(           // fusion layer
    const float* __restrict__ part, const float* __restrict__ bias, int N,
    bf16_t* __restrict__ a_bf, float* __restrict__ ff, bf16_t* __restrict__ ff_bf)
{
    const int idx = blockIdx.x * blockDim.x + threadIdx.x;        // 32*N
    const int r = idx / N, c = idx - r * N;
    float s = bias[c];
    #pragma unroll
    for (int k = 0; k < KSPL; ++k) s += part[((size_t)k * BSZ + r) * N + c];
    const bf16_t h = f2bf(s);
    a_bf[idx] = h; ff[idx] = s; ff_bf[idx] = h;
}

__global__ __launch_bounds__(256) void epi_bn_relu_kernel(        // bias+BN+ReLU
    const float* __restrict__ part, const float* __restrict__ bias,
    const float* __restrict__ g, const float* __restrict__ b, int N,
    bf16_t* __restrict__ out_bf)
{
    const int c = blockIdx.x * blockDim.x + threadIdx.x;          // one column/thread
    float v[BSZ];
    #pragma unroll
    for (int r = 0; r < BSZ; ++r) v[r] = bias[c];
    #pragma unroll
    for (int k = 0; k < KSPL; ++k)
        #pragma unroll
        for (int r = 0; r < BSZ; ++r) v[r] += part[((size_t)k * BSZ + r) * N + c];
    float mu = 0.f;
    #pragma unroll
    for (int r = 0; r < BSZ; ++r) mu += v[r];
    mu *= (1.f / BSZ);
    float var = 0.f;
    #pragma unroll
    for (int r = 0; r < BSZ; ++r) { float d = v[r] - mu; var += d * d; }
    var *= (1.f / BSZ);
    const float rs = rsqrtf(var + BN_EPS);
    const float gg = g[c], bb = b[c];
    #pragma unroll
    for (int r = 0; r < BSZ; ++r)
        out_bf[r * N + c] = f2bf(fmaxf((v[r] - mu) * rs * gg + bb, 0.f));
}

__global__ __launch_bounds__(256) void epi_sigmoid_kernel(
    const float* __restrict__ part, const float* __restrict__ bias, int N,
    float* __restrict__ out)
{
    const int idx = blockIdx.x * blockDim.x + threadIdx.x;
    const int r = idx / N, c = idx - r * N;
    float s = bias[c];
    #pragma unroll
    for (int k = 0; k < KSPL; ++k) s += part[((size_t)k * BSZ + r) * N + c];
    out[idx] = 1.f / (1.f + __expf(-s));
}

__global__ __launch_bounds__(256) void epi_residual_kernel(
    const float* __restrict__ part, const float* __restrict__ bias, int N,
    float* __restrict__ ff, bf16_t* __restrict__ ff_bf)
{
    const int idx = blockIdx.x * blockDim.x + threadIdx.x;
    const int r = idx / N, c = idx - r * N;
    float s = bias[c] + ff[idx];
    #pragma unroll
    for (int k = 0; k < KSPL; ++k) s += part[((size_t)k * BSZ + r) * N + c];
    ff[idx] = s; ff_bf[idx] = f2bf(s);
}

// out[j] identical for all 32 compose calls -> compute once, broadcast
__global__ __launch_bounds__(256) void final_broadcast_kernel(
    const float* __restrict__ img, const float* __restrict__ g,
    const float* __restrict__ ff, float* __restrict__ out)
{
    const int idx = blockIdx.x * blockDim.x + threadIdx.x;        // 32*4096
    const float gv = g[idx];
    const float o = img[idx] * gv + ff[idx] * (1.f - gv);
    #pragma unroll 4
    for (int j = 0; j < BSZ; ++j) out[(size_t)j * BSZ * E_DIM + idx] = o;
}

// ---------------------------------------------------------------------------
extern "C" void kernel_launch(void* const* d_in, const int* in_sizes, int n_in,
                              void* d_out, int out_size, void* d_ws, size_t ws_size,
                              hipStream_t stream)
{
    const float* img     = (const float*)d_in[0];
    const float* fs_bn_g = (const float*)d_in[2];
    const float* fs_bn_b = (const float*)d_in[3];
    const float* fs_w    = (const float*)d_in[4];
    const float* fs_b    = (const float*)d_in[5];
    const float* gt_w1   = (const float*)d_in[6];
    const float* gt_b1   = (const float*)d_in[7];
    const float* gt_bn_g = (const float*)d_in[8];
    const float* gt_bn_b = (const float*)d_in[9];
    const float* gt_w2   = (const float*)d_in[10];
    const float* gt_b2   = (const float*)d_in[11];
    const float* ee_w1   = (const float*)d_in[12];
    const float* ee_b1   = (const float*)d_in[13];
    const float* ee_bn_g = (const float*)d_in[14];
    const float* ee_bn_b = (const float*)d_in[15];
    const float* ee_w2   = (const float*)d_in[16];
    const float* ee_b2   = (const float*)d_in[17];

    char* ws = (char*)d_ws;
    size_t off = 0;
    float*  part  = (float*) (ws + off); off += (size_t)KSPL * BSZ * E_DIM * 4; // 4 MB
    bf16_t* xn    = (bf16_t*)(ws + off); off += (size_t)BSZ * 2 * E_DIM * 2;    // 512 KB
    bf16_t* f_bf  = (bf16_t*)(ws + off); off += (size_t)BSZ * E_DIM * 2;        // 256 KB
    float*  ff    = (float*) (ws + off); off += (size_t)BSZ * E_DIM * 4;        // 512 KB
    bf16_t* ff_bf = (bf16_t*)(ws + off); off += (size_t)BSZ * E_DIM * 2;        // 256 KB
    bf16_t* t1_bf = (bf16_t*)(ws + off); off += (size_t)BSZ * E_DIM * 2;        // 256 KB
    float*  gbuf  = (float*) (ws + off); off += (size_t)BSZ * E_DIM * 4;        // 512 KB
    bf16_t* h_bf  = (bf16_t*)(ws + off); off += (size_t)BSZ * H_DIM * 2;        // 128 KB

    const int elemE = BSZ * E_DIM;                 // 131072
    const dim3 gE(E_DIM / NTB, KSPL);              // (32, 8)
    const dim3 gH(H_DIM / NTB, KSPL);              // (16, 8)

    // xn = relu(BN(concat)) once (text half collapses to beta exactly)
    prep_xn_kernel<<<(2 * E_DIM) / 256, 256, 0, stream>>>(img, fs_bn_g, fs_bn_b, xn);

    // fusion: f = xn @ fs_w + fs_b
    gemm_bf16_partial_kernel<E_DIM><<<gE, 256, 0, stream>>>(xn, fs_w, part, 2 * E_DIM);
    epi_bias_kernel<<<elemE / 256, 256, 0, stream>>>(part, fs_b, E_DIM, f_bf, ff, ff_bf);

    // gating: sigmoid(relu(BN(f@w1+b1)) @ w2 + b2)
    gemm_bf16_partial_kernel<E_DIM><<<gE, 256, 0, stream>>>(f_bf, gt_w1, part, E_DIM);
    epi_bn_relu_kernel<<<E_DIM / 256, 256, 0, stream>>>(part, gt_b1, gt_bn_g, gt_bn_b,
                                                        E_DIM, t1_bf);
    gemm_bf16_partial_kernel<E_DIM><<<gE, 256, 0, stream>>>(t1_bf, gt_w2, part, E_DIM);
    epi_sigmoid_kernel<<<elemE / 256, 256, 0, stream>>>(part, gt_b2, E_DIM, gbuf);

    // 3 error-encoding residual blocks
    for (int i = 0; i < 3; ++i) {
        gemm_bf16_partial_kernel<H_DIM><<<gH, 256, 0, stream>>>(
            ff_bf, ee_w1 + (size_t)i * E_DIM * H_DIM, part, E_DIM);
        epi_bn_relu_kernel<<<H_DIM / 256, 256, 0, stream>>>(
            part, ee_b1 + i * H_DIM, ee_bn_g + i * H_DIM, ee_bn_b + i * H_DIM,
            H_DIM, h_bf);
        gemm_bf16_partial_kernel<E_DIM><<<gE, 256, 0, stream>>>(
            h_bf, ee_w2 + (size_t)i * H_DIM * E_DIM, part, H_DIM);
        epi_residual_kernel<<<elemE / 256, 256, 0, stream>>>(
            part, ee_b2 + i * E_DIM, E_DIM, ff, ff_bf);
    }

    // out[j] = image*g + ff*(1-g), broadcast across 32 compose slices
    final_broadcast_kernel<<<elemE / 256, 256, 0, stream>>>(img, gbuf, ff, (float*)d_out);
}
